// SelfAttention_14482629722560
// MI455X (gfx1250) — matrix operand order repaired and checked by reading, not compiled, not run on hardware
//
#include <hip/hip_runtime.h>
#include <hip/hip_bf16.h>
#include <math.h>

// ---- problem constants (match reference) ----
#define BGR 32      // graphs
#define SEQ 512     // padded seq len
#define DIM 256     // feature dim
#define NH  8       // heads
#define HD  32      // head dim
#define NN  8192    // ragged nodes
#define MS  (BGR*SEQ)   // 16384 dense rows

typedef _Float16 h16;
typedef __attribute__((ext_vector_type(16))) _Float16 v16h;
typedef __attribute__((ext_vector_type(8)))  float    v8f;

// ------------------------------------------------------------------
// WMMA wrapper: D = A(16x32 f16) * B(32x16 f16) + C(16x16 f32)
// ------------------------------------------------------------------
__device__ inline v8f wmma16(v16h a, v16h b, v8f c) {
  return __builtin_amdgcn_wmma_f32_16x16x32_f16(
      false, a, false, b, (short)0, c, false, false);
}

// A fragment (16x32 f16) from row-major f16 memory (global or LDS generic).
// ISA 7.12.2: lanes 0-15 -> M=0..15; VGPR r holds K = (r>>2)*16 + half*8 + (r&3)*2 (+1)
// Per lane this is two contiguous 16-byte chunks -> b128-coalescible.
__device__ inline v16h frag_a(const h16* A, int lda) {
  int lane = threadIdx.x & 31;
  int m = lane & 15, hf = lane >> 4;
  v16h a;
#pragma unroll
  for (int r = 0; r < 8; ++r) {
    int kb = (r >> 2) * 16 + hf * 8 + (r & 3) * 2;
    a[2 * r]     = A[m * lda + kb];
    a[2 * r + 1] = A[m * lda + kb + 1];
  }
  return a;
}

// B fragment (32x16) where B[k][n] = M[n][k]  (multiply by M^T, M row-major).
// Element e holds K = (e>>3)*16 + half*8 + (e&7) (n=lane&15), same order as frag_a.
// Per lane: two contiguous 8-half chunks.
__device__ inline v16h frag_b_T(const h16* M, int ldm) {
  int lane = threadIdx.x & 31;
  int n = lane & 15, hf = lane >> 4;
  v16h b;
#pragma unroll
  for (int e = 0; e < 16; ++e) b[e] = M[n * ldm + (e >> 3) * 16 + hf * 8 + (e & 7)];
  return b;
}

// ------------------------------------------------------------------
// Ragged bookkeeping
// ------------------------------------------------------------------
__global__ void k_init(float* xz, int nwords, int* counts) {
  int i = blockIdx.x * blockDim.x + threadIdx.x;
  for (int k = i; k < nwords; k += gridDim.x * blockDim.x) xz[k] = 0.0f;
  if (i < BGR) counts[i] = 0;
}

__global__ void k_count(const int* batch, int* counts) {
  int i = blockIdx.x * blockDim.x + threadIdx.x;
  if (i < NN) atomicAdd(&counts[batch[i]], 1);
}

__global__ void k_scan(const int* counts, int* starts) {
  if (threadIdx.x == 0) {
    int run = 0;
    for (int b = 0; b < BGR; ++b) { starts[b] = run; run += counts[b]; }
  }
}

__global__ void k_scatter(const float* node, const int* batch, const int* starts, h16* Xh) {
  int idx = blockIdx.x * 256 + threadIdx.x;           // over NN*DIM
  if (idx >= NN * DIM) return;
  int i = idx >> 8, d = idx & 255;
  int b = batch[i];
  int pos = i - starts[b];
  Xh[((b << 9) + pos) * DIM + d] = (h16)node[idx];
}

// all four weight matrices f32 -> f16 in one launch
__global__ void k_cvt4(const float* Wq, const float* Wk, const float* Wv, const float* Wo,
                       h16* Wqh, h16* Wkh, h16* Wvh, h16* Woh) {
  int idx = blockIdx.x * 256 + threadIdx.x;           // over 4*DIM*DIM
  int sel = idx >> 16, i = idx & 0xFFFF;
  const float* s = (sel == 0) ? Wq : (sel == 1) ? Wk : (sel == 2) ? Wv : Wo;
  h16*         d = (sel == 0) ? Wqh : (sel == 1) ? Wkh : (sel == 2) ? Wvh : Woh;
  d[i] = (h16)s[i];
}

// ------------------------------------------------------------------
// Fused QKV projection: Q/K = Xh @ W^T + b (row-major 16384x256),
// V additionally stored TRANSPOSED as Vt[b][h][d][s] so the P*V
// B-operand later becomes contiguous-per-lane (b128 loads).
// 8 waves per block, 1 tile per wave, 3 WMMAs per K-step sharing A.
// ------------------------------------------------------------------
__global__ __launch_bounds__(256) void k_qkv(
    const h16* __restrict__ Xh,
    const h16* __restrict__ Wq, const h16* __restrict__ Wk, const h16* __restrict__ Wv,
    const float* __restrict__ bq, const float* __restrict__ bk, const float* __restrict__ bv,
    h16* __restrict__ Qh, h16* __restrict__ Kh, h16* __restrict__ Vt) {
  int wave = threadIdx.x >> 5;
  int tile = blockIdx.x * 8 + wave;       // 16384 tiles total
  int mt = tile >> 4, nt = tile & 15;
  v8f aq = {}, ak = {}, av = {};
#pragma unroll
  for (int ks = 0; ks < 8; ++ks) {
    v16h A  = frag_a(Xh + (mt * 16) * DIM + ks * 32, DIM);
    v16h Bq = frag_b_T(Wq + (nt * 16) * DIM + ks * 32, DIM);
    v16h Bk = frag_b_T(Wk + (nt * 16) * DIM + ks * 32, DIM);
    v16h Bv = frag_b_T(Wv + (nt * 16) * DIM + ks * 32, DIM);
    aq = wmma16(A, Bq, aq);
    ak = wmma16(A, Bk, ak);
    av = wmma16(A, Bv, av);
  }
  int lane = threadIdx.x & 31;
  int n = lane & 15, mh = (lane >> 4) * 8;
  int col = nt * 16 + n;
  float vbq = bq[col], vbk = bk[col], vbv = bv[col];
  // V transposed store: a 16-row tile never crosses a graph boundary
  int b = (mt * 16) >> 9;                  // graph id
  int s0 = (mt * 16) & 511;                // seq pos of tile row 0
  int hh = col >> 5, dd = col & 31;        // head, head-dim
  h16* vrow = Vt + (size_t)(((b * NH + hh) * HD + dd)) * SEQ + s0 + mh;
#pragma unroll
  for (int r = 0; r < 8; ++r) {
    int row = mt * 16 + mh + r;
    Qh[row * DIM + col] = (h16)(aq[r] + vbq);
    Kh[row * DIM + col] = (h16)(ak[r] + vbk);
    vrow[r]             = (h16)(av[r] + vbv);   // 8 consecutive halves -> b128 store
  }
}

// ------------------------------------------------------------------
// Attention: per (graph, 16-row q-tile) block, loop heads.
// scores (WMMA) -> masked softmax (f32 LDS) -> attn_w into d_out -> P*V (WMMA)
// ------------------------------------------------------------------
__global__ __launch_bounds__(256) void k_attn(
    const h16* __restrict__ Qh, const h16* __restrict__ Kh, const h16* __restrict__ Vt,
    const int* __restrict__ counts,
    float* __restrict__ attnw, h16* __restrict__ Ctxh) {
  __shared__ float s[16 * SEQ];     // 32 KB f32 scores/probs (for attn_w output)
  __shared__ h16   ph[16 * SEQ];    // 16 KB f16 probs (WMMA A operand, contiguous)
  __shared__ float red[256];        // row reductions
  __shared__ float pc[8 * 256];     // 8 KB ctx partials
  int b = blockIdx.x >> 5, qt = blockIdx.x & 31;
  int cnt = counts[b];
  int t = threadIdx.x, wave = t >> 5, lane = t & 31;
  int n = lane & 15, mh = (lane >> 4) * 8;
  const float scale = 0.17677669529663687f;   // 1/sqrt(32)

  for (int h = 0; h < NH; ++h) {
    // ---- scores: 32 key-tiles over 8 waves, one WMMA each (K=HD=32) ----
    v16h Aq = frag_a(Qh + (b * SEQ + qt * 16) * DIM + h * HD, DIM);
    for (int kt = wave; kt < 32; kt += 8) {
      v16h Bk = frag_b_T(Kh + (b * SEQ + kt * 16) * DIM + h * HD, DIM);
      v8f c = {};
      c = wmma16(Aq, Bk, c);
#pragma unroll
      for (int r = 0; r < 8; ++r) {
        int kk = kt * 16 + n;
        float v = c[r] * scale;
        if (kk >= cnt) v = -1e9f;
        s[(mh + r) * SEQ + kk] = v;
      }
    }
    __syncthreads();

    // ---- softmax: 16 threads per row, 32 contiguous elems per thread ----
    int row = t >> 4, cch = t & 15;
    float* srow = s + row * SEQ + cch * 32;
    h16*   prow = ph + row * SEQ + cch * 32;
    float mx = -3.0e38f;
#pragma unroll
    for (int j = 0; j < 32; ++j) mx = fmaxf(mx, srow[j]);
    red[t] = mx; __syncthreads();
    for (int off = 8; off > 0; off >>= 1) {
      if (cch < off) red[t] = fmaxf(red[t], red[t + off]);
      __syncthreads();
    }
    float rmax = red[row * 16]; __syncthreads();
    float sm = 0.0f;
#pragma unroll
    for (int j = 0; j < 32; ++j) { float e = __expf(srow[j] - rmax); srow[j] = e; sm += e; }
    red[t] = sm; __syncthreads();
    for (int off = 8; off > 0; off >>= 1) {
      if (cch < off) red[t] += red[t + off];
      __syncthreads();
    }
    float inv = 1.0f / red[row * 16]; __syncthreads();
#pragma unroll
    for (int j = 0; j < 32; ++j) {
      float p = srow[j] * inv;
      srow[j] = p;               // f32 copy for attn_w
      prow[j] = (h16)p;          // f16 copy for WMMA A operand
    }
    __syncthreads();

    // ---- head-averaged attention straight into d_out ----
    for (int idx = t; idx < 16 * SEQ; idx += 256) {
      long oi = ((long)(b * SEQ + qt * 16 + (idx >> 9))) * SEQ + (idx & 511);
      float v = s[idx] * (1.0f / NH);
      if (h == 0) attnw[oi] = v; else attnw[oi] += v;
    }

    // ---- ctx = P @ V : K=512 split 4 ways per n-half, partials in LDS ----
    // B operand from Vt[b][h][d][s]: b[e] = Vt[...][kt*32 + (e>>3)*16 + hf*8 + (e&7)] (two 8-half runs)
    int nh = wave & 1, kq = wave >> 1;
    const h16* vbase = Vt + (size_t)((b * NH + h) * HD + nh * 16) * SEQ;
    v8f acc = {};
#pragma unroll
    for (int i2 = 0; i2 < 4; ++i2) {
      int kt = kq * 4 + i2;                       // 32-key chunk
      v16h Ap = frag_a(ph + kt * 32, SEQ);        // f16 probs from LDS
      v16h Bv = frag_b_T(vbase + kt * 32, SEQ);   // contiguous V^T rows
      acc = wmma16(Ap, Bv, acc);
    }
#pragma unroll
    for (int r = 0; r < 8; ++r) pc[wave * 256 + (mh + r) * 16 + n] = acc[r];
    __syncthreads();
    for (int it = 0; it < 2; ++it) {
      int idx = t + it * 256;                     // 0..511 = 16x32 outputs
      int m = idx >> 5, col = idx & 31;
      int nh2 = col >> 4, nn = col & 15;
      float sum = pc[(nh2) * 256 + m * 16 + nn]
                + pc[(nh2 + 2) * 256 + m * 16 + nn]
                + pc[(nh2 + 4) * 256 + m * 16 + nn]
                + pc[(nh2 + 6) * 256 + m * 16 + nn];
      Ctxh[(long)(b * SEQ + qt * 16 + m) * DIM + h * HD + col] = (h16)sum;
    }
    __syncthreads();
  }
}

// ------------------------------------------------------------------
// Output projection + LayerNorm + ragged gather/residual
// One block per 16-row tile; 8 waves x 2 n-tiles each; LN in f32 LDS.
// ------------------------------------------------------------------
__global__ __launch_bounds__(256) void k_out(
    const h16* __restrict__ Ctxh, const h16* __restrict__ Wo,
    const float* __restrict__ bo, const float* __restrict__ gamma,
    const float* __restrict__ beta, const float* __restrict__ node,
    const int* __restrict__ counts, const int* __restrict__ starts,
    float* __restrict__ nodeout) {
  __shared__ float o[16 * DIM];     // 16 KB
  __shared__ float red[256];
  int m0 = blockIdx.x * 16;
  int t = threadIdx.x, wave = t >> 5, lane = t & 31;
  int n = lane & 15, mh = (lane >> 4) * 8;

  for (int w2 = 0; w2 < 2; ++w2) {
    int nt = wave * 2 + w2;
    v8f acc = {};
#pragma unroll
    for (int ks = 0; ks < 8; ++ks) {
      v16h A = frag_a(Ctxh + m0 * DIM + ks * 32, DIM);
      v16h B = frag_b_T(Wo + (nt * 16) * DIM + ks * 32, DIM);
      acc = wmma16(A, B, acc);
    }
    float bv = bo[nt * 16 + n];
#pragma unroll
    for (int r = 0; r < 8; ++r) o[(mh + r) * DIM + nt * 16 + n] = acc[r] + bv;
  }
  __syncthreads();

  // LayerNorm: 16 threads per row, 16 cols per thread
  int row = t >> 4, c = t & 15;
  float* orow = o + row * DIM + c * 16;
  float sm = 0.0f;
#pragma unroll
  for (int j = 0; j < 16; ++j) sm += orow[j];
  red[t] = sm; __syncthreads();
  for (int off = 8; off > 0; off >>= 1) {
    if (c < off) red[t] += red[t + off];
    __syncthreads();
  }
  float mu = red[row * 16] * (1.0f / DIM); __syncthreads();
  float vs = 0.0f;
#pragma unroll
  for (int j = 0; j < 16; ++j) { float d = orow[j] - mu; vs += d * d; }
  red[t] = vs; __syncthreads();
  for (int off = 8; off > 0; off >>= 1) {
    if (c < off) red[t] += red[t + off];
    __syncthreads();
  }
  float var = red[row * 16] * (1.0f / DIM);
  float rstd = rsqrtf(var + 1e-5f);

  int m = m0 + row, b = m >> 9, posr = m & 511;
  if (posr < counts[b]) {
    long i = (long)starts[b] + posr;
#pragma unroll
    for (int j = 0; j < 16; ++j) {
      int col = c * 16 + j;
      nodeout[i * DIM + col] =
          (orow[j] - mu) * rstd * gamma[col] + beta[col] + node[i * DIM + col];
    }
  }
}

// ------------------------------------------------------------------
extern "C" void kernel_launch(void* const* d_in, const int* in_sizes, int n_in,
                              void* d_out, int out_size, void* d_ws, size_t ws_size,
                              hipStream_t stream) {
  const float* node  = (const float*)d_in[0];
  const int*   batch = (const int*)  d_in[1];
  const float* Wq    = (const float*)d_in[2];
  const float* Wk    = (const float*)d_in[3];
  const float* Wv    = (const float*)d_in[4];
  const float* Wo    = (const float*)d_in[5];
  const float* bq    = (const float*)d_in[6];
  const float* bk    = (const float*)d_in[7];
  const float* bv    = (const float*)d_in[8];
  const float* bo    = (const float*)d_in[9];
  const float* gamma = (const float*)d_in[10];
  const float* beta  = (const float*)d_in[11];

  float* nodeout = (float*)d_out;                 // NN*DIM
  float* attnw   = nodeout + (size_t)NN * DIM;    // BGR*SEQ*SEQ

  // workspace layout
  char* w = (char*)d_ws;
  size_t off = 0;
  int* counts = (int*)(w + off); off += 128;
  int* starts = (int*)(w + off); off += 128;
  off = (off + 255) & ~(size_t)255;
  h16* Wqh = (h16*)(w + off); off += (size_t)DIM * DIM * 2;
  h16* Wkh = (h16*)(w + off); off += (size_t)DIM * DIM * 2;
  h16* Wvh = (h16*)(w + off); off += (size_t)DIM * DIM * 2;
  h16* Woh = (h16*)(w + off); off += (size_t)DIM * DIM * 2;
  h16* Xh  = (h16*)(w + off); off += (size_t)MS * DIM * 2;
  h16* Qh  = (h16*)(w + off); off += (size_t)MS * DIM * 2;
  h16* Kh  = (h16*)(w + off); off += (size_t)MS * DIM * 2;
  h16* Vt  = (h16*)(w + off); off += (size_t)MS * DIM * 2;  // [b][h][d][s]
  h16* Ctx = (h16*)(w + off); off += (size_t)MS * DIM * 2;

  // 1) zero dense staging + counts
  k_init<<<2048, 256, 0, stream>>>((float*)Xh, MS * DIM / 2, counts);
  // 2) histogram + exclusive scan -> starts
  k_count<<<(NN + 255) / 256, 256, 0, stream>>>(batch, counts);
  k_scan<<<1, 32, 0, stream>>>(counts, starts);
  // 3) scatter ragged -> dense f16
  k_scatter<<<(NN * DIM) / 256, 256, 0, stream>>>(node, batch, starts, Xh);
  // 4) weights -> f16 (one launch)
  k_cvt4<<<(4 * DIM * DIM) / 256, 256, 0, stream>>>(Wq, Wk, Wv, Wo, Wqh, Wkh, Wvh, Woh);
  // 5) QKV projection (WMMA), V stored transposed
  k_qkv<<<(MS / 16) * 16 / 8, 256, 0, stream>>>(Xh, Wqh, Wkh, Wvh, bq, bk, bv, Qh, Kh, Vt);
  // 6) attention + softmax + attn_w + P*V (WMMA)
  k_attn<<<BGR * (SEQ / 16), 256, 0, stream>>>(Qh, Kh, Vt, counts, attnw, Ctx);
  // 7) out projection + LayerNorm + ragged residual gather (WMMA)
  k_out<<<MS / 16, 256, 0, stream>>>(Ctx, Woh, bo, gamma, beta, node, counts, starts, nodeout);
}
